// KineticsGNN_62732292326002
// MI455X (gfx1250) — matrix operand order repaired and checked by reading, not compiled, not run on hardware
//
#include <hip/hip_runtime.h>

// ---------------- problem constants (match reference) ----------------
#define NN 100000   // nodes
#define EE 1600000  // directed edges
#define FF 22       // input features
#define FP 32       // padded input features (multiple of WMMA K=32)
#define HH 128      // hidden dim
#define GG 256      // graphs
#define HD 64       // head hidden
#define BN_EPS 1e-5f

#define MTILE 4                         // row tiles per wave (B-fragment reuse)
#define RT_TOT (NN / 16)                // 6250 row tiles
#define RT_PAD ((RT_TOT + MTILE - 1) / MTILE * MTILE)  // 6252 (padded for MT blocking)

typedef _Float16 v16h __attribute__((ext_vector_type(16)));
typedef float    v8f  __attribute__((ext_vector_type(8)));

__device__ __forceinline__ void atomicAddF(float* p, float v) {
  __hip_atomic_fetch_add(p, v, __ATOMIC_RELAXED, __HIP_MEMORY_SCOPE_AGENT);
}

// ---- fragment-layout index helpers (CDNA5 ISA 7.12.2, 16-bit A 16x32 / B 32x16) ----
// A element (n, f): fragment ((rt*KC+kc)*32 + lane) holds 16 halves; within a
// 32-wide K chunk, K = 2r + (lane_hi?8:0) + odd (r<4) or 16 + 2(r-4) + (lane_hi?8:0) + odd.
__device__ __forceinline__ size_t a_pack_idx(int n, int f, int KC) {
  int rt = n >> 4, lm = n & 15;
  int kc = f >> 5, q = f & 31;
  int base16 = (q >= 16) ? 1 : 0;
  int q2 = q & 15;
  int l_hi = (q2 >= 8) ? 1 : 0;
  int t = q2 & 7;
  int r = (t >> 1) + base16 * 4;
  int h = 2 * r + (t & 1);
  int lane = l_hi * 16 + lm;
  return ((size_t)(rt * KC + kc) * 32 + lane) * 16 + h;   // units: halves
}
// B element (k, n): lane = N%16 (+16 for (K%16)>=8 within chunk), half = K%8 (+8 for K>=16).
__device__ __forceinline__ size_t b_pack_idx(int k, int n, int KC) {
  int ct = n >> 4, lm = n & 15;
  int kc = k >> 5, q = k & 31;
  int l_hi = ((q & 15) >= 8) ? 1 : 0;
  int h = (q & 7) + ((q >= 16) ? 8 : 0);
  int lane = l_hi * 16 + lm;
  return ((size_t)(ct * KC + kc) * 32 + lane) * 16 + h;
}

// ---------------- tiny utility kernels ----------------
__global__ void k_fill_f32(float* __restrict__ p, int n, float v) {
  int i = blockIdx.x * blockDim.x + threadIdx.x;
  if (i < n) p[i] = v;
}

__global__ void k_degree(const int* __restrict__ col, float* __restrict__ deg) {
  int e = blockIdx.x * blockDim.x + threadIdx.x;
  if (e < EE) atomicAddF(&deg[col[e]], 1.0f);
}

__global__ void k_rsqrt(const float* __restrict__ deg, float* __restrict__ dinv) {
  int i = blockIdx.x * blockDim.x + threadIdx.x;
  if (i < NN) dinv[i] = (deg[i] > 0.f) ? rsqrtf(deg[i]) : 0.f;
}

// x [N,22] f32 -> packed A fragments [N,32] f16 (zero-padded K)
__global__ void k_pack_x(const float* __restrict__ x, _Float16* __restrict__ xp) {
  int i = blockIdx.x * blockDim.x + threadIdx.x;
  if (i >= NN * FP) return;
  int n = i >> 5, f = i & 31;
  float v = (f < FF) ? x[n * FF + f] : 0.f;
  xp[a_pack_idx(n, f, 1)] = (_Float16)v;
}

// W [srcK, cols] f32 -> packed B fragments [Kpad, cols] f16 (zero-padded K)
__global__ void k_pack_w(const float* __restrict__ w, _Float16* __restrict__ wp,
                         int srcK, int Kpad, int cols) {
  int i = blockIdx.x * blockDim.x + threadIdx.x;
  if (i >= Kpad * cols) return;
  int k = i / cols, n = i % cols;
  float v = (k < srcK) ? w[k * cols + n] : 0.f;
  wp[b_pack_idx(k, n, Kpad / 32)] = (_Float16)v;
}

// ---------------- WMMA f16->f32 GEMM ----------------
// out[M,Hout] = A[M,K] @ W[K,Hout]; A/B pre-packed in fragment layout.
// grid.x = ceil(Mtiles/MT); block.x = 32*(Hout/16). Block stages its A
// fragments through LDS once (coalesced), all col-tile waves reuse them.
template <int MT, int KC, bool BIAS, bool RELU>
__global__ void k_gemm_wmma(const v16h* __restrict__ Ap, const v16h* __restrict__ Bp,
                            const float* __restrict__ bias, float* __restrict__ out,
                            int Mtiles, int Hout) {
  __shared__ v16h As[MT * KC * 32];
  const int tid  = threadIdx.x;
  const int lane = tid & 31;
  const int ct   = tid >> 5;          // column tile = wave id
  const int rt0  = blockIdx.x * MT;

  // coalesced block-wide copy of A fragments (global -> LDS)
  const v16h* Ablk = Ap + (size_t)rt0 * KC * 32;
  for (int idx = tid; idx < MT * KC * 32; idx += blockDim.x) As[idx] = Ablk[idx];
  __syncthreads();

  v8f c[MT];
#pragma unroll
  for (int t = 0; t < MT; ++t)
#pragma unroll
    for (int r = 0; r < 8; ++r) c[t][r] = 0.f;

#pragma unroll
  for (int kc = 0; kc < KC; ++kc) {
    v16h b = Bp[(size_t)(ct * KC + kc) * 32 + lane];    // 2x global_load_b128
#pragma unroll
    for (int t = 0; t < MT; ++t) {
      v16h a = As[(t * KC + kc) * 32 + lane];           // 2x ds_load_b128
      c[t] = __builtin_amdgcn_wmma_f32_16x16x32_f16(false, a, false, b, (short)0,
                                                    c[t], false, false);
    }
  }

  const int col   = ct * 16 + (lane & 15);
  const int l_hi8 = (lane >> 4) * 8;
  const float bv  = BIAS ? bias[col] : 0.f;
#pragma unroll
  for (int t = 0; t < MT; ++t) {
    int rt = rt0 + t;
    if (rt < Mtiles) {
#pragma unroll
      for (int r = 0; r < 8; ++r) {
        float v = c[t][r] + bv;
        if (RELU) v = fmaxf(v, 0.f);
        out[(size_t)(rt * 16 + r + l_hi8) * Hout + col] = v;
      }
    }
  }
}

// ---------------- GCN propagation ----------------
// acc[n,f] = b[f] + dinv[n]^2 * hW[n,f]   (self-loop + bias; full overwrite)
__global__ void k_scatter_init(const float* __restrict__ hW, const float* __restrict__ dinv,
                               const float* __restrict__ bias, float* __restrict__ acc) {
  long long i = (long long)blockIdx.x * blockDim.x + threadIdx.x;
  if (i >= (long long)NN * HH) return;
  int n = (int)(i >> 7), f = (int)(i & 127);
  float d = dinv[n];
  acc[i] = bias[f] + d * d * hW[i];
}

// acc[col[e],f] += dinv[row]*dinv[col] * hW[row[e],f]   (one wave per edge)
__global__ void k_scatter_edges(const int* __restrict__ row, const int* __restrict__ col,
                                const float* __restrict__ dinv, const float* __restrict__ hW,
                                float* __restrict__ acc) {
  long long tid = (long long)blockIdx.x * blockDim.x + threadIdx.x;
  if (tid >= (long long)EE * 32) return;
  int e = (int)(tid >> 5);
  int lane = (int)(tid & 31);
  int r = row[e], c = col[e];               // wave-uniform -> scalarized
  float w = dinv[r] * dinv[c];
  const float* src = hW + (long long)r * HH;
  float* dst = acc + (long long)c * HH;
#pragma unroll
  for (int j = 0; j < 4; ++j) {
    int f = lane + 32 * j;                  // coalesced across the wave
    atomicAddF(&dst[f], w * src[f]);
  }
}

// hp (packed A fragments) = (f16) relu( (acc-m) * g*rsqrt(v+eps) + be )
__global__ void k_bn_relu_pack(const float* __restrict__ acc, const float* __restrict__ g,
                               const float* __restrict__ be, const float* __restrict__ m,
                               const float* __restrict__ v, _Float16* __restrict__ hp) {
  long long i = (long long)blockIdx.x * blockDim.x + threadIdx.x;
  if (i >= (long long)NN * HH) return;
  int n = (int)(i >> 7), f = (int)(i & 127);
  float y = (acc[i] - m[f]) * (g[f] * rsqrtf(v[f] + BN_EPS)) + be[f];
  hp[a_pack_idx(n, f, HH / 32)] = (_Float16)fmaxf(y, 0.f);
}

// ---------------- pooling + heads ----------------
__global__ void k_pool(const _Float16* __restrict__ hp, const int* __restrict__ batch,
                       float* __restrict__ sums, float* __restrict__ cnt) {
  long long i = (long long)blockIdx.x * blockDim.x + threadIdx.x;
  if (i >= (long long)NN * HH) return;
  int n = (int)(i >> 7), f = (int)(i & 127);
  int b = batch[n];
  atomicAddF(&sums[b * HH + f], (float)hp[a_pack_idx(n, f, HH / 32)]);
  if (f == 0) atomicAddF(&cnt[b], 1.0f);
}

__global__ void k_pool_div_pack(const float* __restrict__ sums, const float* __restrict__ cnt,
                                _Float16* __restrict__ poolp) {
  int i = blockIdx.x * blockDim.x + threadIdx.x;
  if (i >= GG * HH) return;
  int g = i >> 7, f = i & 127;
  poolp[a_pack_idx(g, f, HH / 32)] = (_Float16)(sums[i] / fmaxf(cnt[g], 1.0f));
}

__global__ void k_heads_final(const float* __restrict__ hk, const float* __restrict__ hm,
                              const float* __restrict__ Wk2, const float* __restrict__ bk2,
                              const float* __restrict__ Wm2, const float* __restrict__ bm2,
                              float* __restrict__ out) {
  int t = blockIdx.x * blockDim.x + threadIdx.x;
  if (t < GG) {
    float s = bk2[0];
#pragma unroll 8
    for (int j = 0; j < HD; ++j) s += hk[t * HD + j] * Wk2[j];
    out[t] = s;
  } else if (t < 2 * GG) {
    int g = t - GG;
    float s = bm2[0];
#pragma unroll 8
    for (int j = 0; j < HD; ++j) s += hm[g * HD + j] * Wm2[j];
    out[GG + g] = s;
  }
}

// ---------------- host-side orchestration ----------------
static inline int cdiv(long long a, int b) { return (int)((a + b - 1) / b); }

extern "C" void kernel_launch(void* const* d_in, const int* in_sizes, int n_in,
                              void* d_out, int out_size, void* d_ws, size_t ws_size,
                              hipStream_t stream) {
  const float* x     = (const float*)d_in[0];
  const int*   edges = (const int*)d_in[1];           // [2,E] flat: row then col
  const int*   batch = (const int*)d_in[2];
  const float* W1 = (const float*)d_in[3];  const float* b1 = (const float*)d_in[4];
  const float* W2 = (const float*)d_in[5];  const float* b2 = (const float*)d_in[6];
  const float* W3 = (const float*)d_in[7];  const float* b3 = (const float*)d_in[8];
  const float* g1 = (const float*)d_in[9];  const float* be1 = (const float*)d_in[10];
  const float* m1 = (const float*)d_in[11]; const float* v1  = (const float*)d_in[12];
  const float* g2 = (const float*)d_in[13]; const float* be2 = (const float*)d_in[14];
  const float* m2 = (const float*)d_in[15]; const float* v2  = (const float*)d_in[16];
  const float* g3 = (const float*)d_in[17]; const float* be3 = (const float*)d_in[18];
  const float* m3 = (const float*)d_in[19]; const float* v3  = (const float*)d_in[20];
  const float* Wk1 = (const float*)d_in[21]; const float* bk1 = (const float*)d_in[22];
  const float* Wk2 = (const float*)d_in[23]; const float* bk2 = (const float*)d_in[24];
  const float* Wm1 = (const float*)d_in[25]; const float* bm1 = (const float*)d_in[26];
  const float* Wm2 = (const float*)d_in[27]; const float* bm2 = (const float*)d_in[28];

  const int* row = edges;
  const int* col = edges + EE;

  // workspace carve-out (~136 MB)
  char* ws = (char*)d_ws;
  size_t off = 0;
  auto alloc = [&](size_t bytes) -> void* {
    void* p = ws + off;
    off = (off + bytes + 255) & ~(size_t)255;
    return p;
  };
  float*    deg   = (float*)alloc((size_t)NN * 4);
  float*    dinv  = (float*)alloc((size_t)NN * 4);
  _Float16* xp    = (_Float16*)alloc((size_t)RT_PAD * 1 * 512 * 2);   // packed A, KC=1
  _Float16* hp    = (_Float16*)alloc((size_t)RT_PAD * 4 * 512 * 2);   // packed A, KC=4
  float*    hW    = (float*)alloc((size_t)NN * HH * 4);
  float*    acc   = (float*)alloc((size_t)NN * HH * 4);
  _Float16* W1p   = (_Float16*)alloc((size_t)(HH / 16) * 1 * 512 * 2);
  _Float16* W2p   = (_Float16*)alloc((size_t)(HH / 16) * 4 * 512 * 2);
  _Float16* W3p   = (_Float16*)alloc((size_t)(HH / 16) * 4 * 512 * 2);
  _Float16* Wk1p  = (_Float16*)alloc((size_t)(HD / 16) * 4 * 512 * 2);
  _Float16* Wm1p  = (_Float16*)alloc((size_t)(HD / 16) * 4 * 512 * 2);
  float*    sums  = (float*)alloc((size_t)GG * HH * 4);
  float*    cnt   = (float*)alloc((size_t)GG * 4);
  _Float16* poolp = (_Float16*)alloc((size_t)(GG / 16) * 4 * 512 * 2);
  float*    hk    = (float*)alloc((size_t)GG * HD * 4);
  float*    hm    = (float*)alloc((size_t)GG * HD * 4);
  (void)ws_size; (void)in_sizes; (void)n_in; (void)out_size;

  const int B = 256;
  const long long NH = (long long)NN * HH;

  // degrees / symmetric normalization
  k_fill_f32<<<cdiv(NN, B), B, 0, stream>>>(deg, NN, 1.0f);   // self loop
  k_degree<<<cdiv(EE, B), B, 0, stream>>>(col, deg);
  k_rsqrt<<<cdiv(NN, B), B, 0, stream>>>(deg, dinv);

  // pack operands into WMMA fragment layout (one-time streaming kernels)
  k_pack_x<<<cdiv((long long)NN * FP, B), B, 0, stream>>>(x, xp);
  k_pack_w<<<cdiv(FP * HH, B), B, 0, stream>>>(W1, W1p, FF, FP, HH);
  k_pack_w<<<cdiv(HH * HH, B), B, 0, stream>>>(W2, W2p, HH, HH, HH);
  k_pack_w<<<cdiv(HH * HH, B), B, 0, stream>>>(W3, W3p, HH, HH, HH);
  k_pack_w<<<cdiv(HH * HD, B), B, 0, stream>>>(Wk1, Wk1p, HH, HH, HD);
  k_pack_w<<<cdiv(HH * HD, B), B, 0, stream>>>(Wm1, Wm1p, HH, HH, HD);

  const int gemmGrid  = RT_PAD / MTILE;   // 1563
  const int gemmBlock = 32 * (HH / 16);   // 8 waves = 8 col tiles
  const long long edgeThreads = (long long)EE * 32;

  // ---- layer 1 ----
  k_gemm_wmma<MTILE, 1, false, false><<<gemmGrid, gemmBlock, 0, stream>>>(
      (const v16h*)xp, (const v16h*)W1p, nullptr, hW, RT_TOT, HH);
  k_scatter_init<<<cdiv(NH, B), B, 0, stream>>>(hW, dinv, b1, acc);
  k_scatter_edges<<<cdiv(edgeThreads, B), B, 0, stream>>>(row, col, dinv, hW, acc);
  k_bn_relu_pack<<<cdiv(NH, B), B, 0, stream>>>(acc, g1, be1, m1, v1, hp);

  // ---- layer 2 ----
  k_gemm_wmma<MTILE, 4, false, false><<<gemmGrid, gemmBlock, 0, stream>>>(
      (const v16h*)hp, (const v16h*)W2p, nullptr, hW, RT_TOT, HH);
  k_scatter_init<<<cdiv(NH, B), B, 0, stream>>>(hW, dinv, b2, acc);
  k_scatter_edges<<<cdiv(edgeThreads, B), B, 0, stream>>>(row, col, dinv, hW, acc);
  k_bn_relu_pack<<<cdiv(NH, B), B, 0, stream>>>(acc, g2, be2, m2, v2, hp);

  // ---- layer 3 ----
  k_gemm_wmma<MTILE, 4, false, false><<<gemmGrid, gemmBlock, 0, stream>>>(
      (const v16h*)hp, (const v16h*)W3p, nullptr, hW, RT_TOT, HH);
  k_scatter_init<<<cdiv(NH, B), B, 0, stream>>>(hW, dinv, b3, acc);
  k_scatter_edges<<<cdiv(edgeThreads, B), B, 0, stream>>>(row, col, dinv, hW, acc);
  k_bn_relu_pack<<<cdiv(NH, B), B, 0, stream>>>(acc, g3, be3, m3, v3, hp);

  // ---- global mean pool ----
  k_fill_f32<<<cdiv(GG * HH, B), B, 0, stream>>>(sums, GG * HH, 0.f);
  k_fill_f32<<<cdiv(GG, B), B, 0, stream>>>(cnt, GG, 0.f);
  k_pool<<<cdiv(NH, B), B, 0, stream>>>(hp, batch, sums, cnt);
  k_pool_div_pack<<<cdiv(GG * HH, B), B, 0, stream>>>(sums, cnt, poolp);

  // ---- heads (fused bias+relu in WMMA epilogue) ----
  k_gemm_wmma<MTILE, 4, true, true><<<(GG / 16) / MTILE, 32 * (HD / 16), 0, stream>>>(
      (const v16h*)poolp, (const v16h*)Wk1p, bk1, hk, GG / 16, HD);
  k_gemm_wmma<MTILE, 4, true, true><<<(GG / 16) / MTILE, 32 * (HD / 16), 0, stream>>>(
      (const v16h*)poolp, (const v16h*)Wm1p, bm1, hm, GG / 16, HD);
  k_heads_final<<<2, 256, 0, stream>>>(hk, hm, Wk2, bk2, Wm2, bm2, (float*)d_out);
}
